// QLoRALinear_26173530702087
// MI455X (gfx1250) — matrix operand order repaired and checked by reading, not compiled, not run on hardware
//
#include <hip/hip_runtime.h>
#include <hip/hip_bf16.h>

typedef __attribute__((ext_vector_type(16))) _Float16 v16h;
typedef __attribute__((ext_vector_type(8)))  float    v8f;
typedef __attribute__((ext_vector_type(4)))  int      v4i;

// Problem constants (from reference setup_inputs)
constexpr int M_TOTAL = 4 * 2048;   // B * S
constexpr int K_IN    = 4096;
constexpr int N_OUT   = 4096;
constexpr int R_LORA  = 16;

// Tiling
constexpr int BM = 128;
constexpr int BN = 128;
constexpr int BK = 64;
constexpr int LDH = 72;             // padded LDS row stride (halves), 144B = 9*16B

struct __align__(8) H4 { _Float16 h[4]; };
struct __align__(16) H8 { _Float16 h[8]; };
union AF { uint4 u[2]; v16h v; };   // 32-byte fragment (8 VGPRs)

// ---------------- CDNA5 async-to-LDS feature gate ----------------
#if defined(__has_builtin)
#if __has_builtin(__builtin_amdgcn_global_load_async_to_lds_b128)
#define HAVE_ASYNC_LDS 1
#endif
#endif
#ifndef HAVE_ASYNC_LDS
#define HAVE_ASYNC_LDS 0
#endif

typedef __attribute__((address_space(1))) v4i* gv4i_ptr;
typedef __attribute__((address_space(3))) v4i* lv4i_ptr;

__device__ __forceinline__ void async_copy_b128(const void* gsrc, void* ldst) {
#if HAVE_ASYNC_LDS
  __builtin_amdgcn_global_load_async_to_lds_b128(
      (gv4i_ptr)gsrc, (lv4i_ptr)ldst, 0, 0);
#else
  *reinterpret_cast<uint4*>(ldst) = *reinterpret_cast<const uint4*>(gsrc);
#endif
}

template <int N>
__device__ __forceinline__ void wait_asynccnt_le() {
#if HAVE_ASYNC_LDS
#if __has_builtin(__builtin_amdgcn_s_wait_asynccnt)
  __builtin_amdgcn_s_wait_asynccnt((short)N);
#else
  asm volatile("s_wait_asynccnt %0" ::"n"(N));
#endif
#endif
}

// --------------------------------------------------------------------------
// Pass 0: x (fp32) -> Xh (f16), 8 elements per thread
// --------------------------------------------------------------------------
__global__ __launch_bounds__(256) void cvt_x_kernel(
    const float* __restrict__ x, _Float16* __restrict__ Xh)
{
  size_t base = ((size_t)blockIdx.x * 256 + threadIdx.x) * 8;
  float4 a = *reinterpret_cast<const float4*>(x + base);
  float4 b = *reinterpret_cast<const float4*>(x + base + 4);
  H8 h;
  h.h[0] = (_Float16)a.x; h.h[1] = (_Float16)a.y;
  h.h[2] = (_Float16)a.z; h.h[3] = (_Float16)a.w;
  h.h[4] = (_Float16)b.x; h.h[5] = (_Float16)b.y;
  h.h[6] = (_Float16)b.z; h.h[7] = (_Float16)b.w;
  *reinterpret_cast<H8*>(Xh + base) = h;
}

// --------------------------------------------------------------------------
// Pass 1: dequant int4 codes -> Wh (f16); one 16-element group per thread
// --------------------------------------------------------------------------
__global__ __launch_bounds__(256) void dequant_w_kernel(
    const int* __restrict__ qw, const float* __restrict__ scale,
    _Float16* __restrict__ Wh)
{
  constexpr float DQ = 2.0f / 15.0f;
  size_t gid  = (size_t)blockIdx.x * 256 + threadIdx.x;  // group id
  size_t base = gid * 16;
  float  s    = scale[gid];
  #pragma unroll
  for (int half = 0; half < 2; ++half) {
    int4 q0 = *reinterpret_cast<const int4*>(qw + base + half * 8);
    int4 q1 = *reinterpret_cast<const int4*>(qw + base + half * 8 + 4);
    H8 h;
    h.h[0] = (_Float16)(fmaf((float)q0.x, DQ, -1.0f) * s);
    h.h[1] = (_Float16)(fmaf((float)q0.y, DQ, -1.0f) * s);
    h.h[2] = (_Float16)(fmaf((float)q0.z, DQ, -1.0f) * s);
    h.h[3] = (_Float16)(fmaf((float)q0.w, DQ, -1.0f) * s);
    h.h[4] = (_Float16)(fmaf((float)q1.x, DQ, -1.0f) * s);
    h.h[5] = (_Float16)(fmaf((float)q1.y, DQ, -1.0f) * s);
    h.h[6] = (_Float16)(fmaf((float)q1.z, DQ, -1.0f) * s);
    h.h[7] = (_Float16)(fmaf((float)q1.w, DQ, -1.0f) * s);
    *reinterpret_cast<H8*>(Wh + base + half * 8) = h;
  }
}

// --------------------------------------------------------------------------
// Kernel 1: T[m][r] = sum_k x[m][k] * lora_A[r][k]    (T in workspace, fp32)
// --------------------------------------------------------------------------
__global__ __launch_bounds__(256) void lora_xa_kernel(
    const float* __restrict__ x, const float* __restrict__ lA,
    float* __restrict__ T)
{
  __shared__ float Xc[128][68];
  __shared__ float Ac[16][68];

  const int t  = threadIdx.x;
  const int m0 = blockIdx.x * 128;
  const int m  = t & 127;
  const int r0 = (t >> 7) * 8;

  float acc[8];
  #pragma unroll
  for (int i = 0; i < 8; ++i) acc[i] = 0.f;

  for (int kc = 0; kc < K_IN / 64; ++kc) {
    const int k0 = kc * 64;
    #pragma unroll
    for (int it = 0; it < 8; ++it) {
      int idx = it * 256 + t;
      int row = idx >> 4;
      int c4  = (idx & 15) * 4;
      float4 v = *reinterpret_cast<const float4*>(
          x + (size_t)(m0 + row) * K_IN + k0 + c4);
      *reinterpret_cast<float4*>(&Xc[row][c4]) = v;
    }
    {
      int row = t >> 4;
      int c4  = (t & 15) * 4;
      float4 v = *reinterpret_cast<const float4*>(
          lA + (size_t)row * K_IN + k0 + c4);
      *reinterpret_cast<float4*>(&Ac[row][c4]) = v;
    }
    __syncthreads();

    #pragma unroll 4
    for (int k = 0; k < 64; ++k) {
      float xv = Xc[m][k];
      #pragma unroll
      for (int rr = 0; rr < 8; ++rr)
        acc[rr] = fmaf(xv, Ac[r0 + rr][k], acc[rr]);
    }
    __syncthreads();
  }

  #pragma unroll
  for (int rr = 0; rr < 8; ++rr)
    T[(size_t)(m0 + m) * R_LORA + r0 + rr] = acc[rr];
}

// ==========================================================================
// Shared epilogue helper: bias + rank-16 LoRA + store
// ==========================================================================
__device__ __forceinline__ void gemm_epilogue(
    v8f (&acc)[2][4], float (&Tl)[BM][16], float (&Bl)[BN][16],
    const float* __restrict__ T, const float* __restrict__ lB,
    const float* __restrict__ bias, float* __restrict__ out,
    int t, int m0, int n0, int wm, int wn, int lr, int half)
{
  __syncthreads();
  #pragma unroll
  for (int it = 0; it < 2; ++it) {
    int idx = it * 256 + t;
    int row = idx >> 2;
    int r4  = (idx & 3) * 4;
    *reinterpret_cast<float4*>(&Tl[row][r4]) =
        *reinterpret_cast<const float4*>(T  + (size_t)(m0 + row) * R_LORA + r4);
    *reinterpret_cast<float4*>(&Bl[row][r4]) =
        *reinterpret_cast<const float4*>(lB + (size_t)(n0 + row) * R_LORA + r4);
  }
  __syncthreads();

  #pragma unroll
  for (int mi = 0; mi < 2; ++mi) {
    #pragma unroll
    for (int ni = 0; ni < 4; ++ni) {
      int colL = wn * 64 + ni * 16 + lr;
      int gcol = n0 + colL;
      float bv = bias[gcol];
      #pragma unroll
      for (int j = 0; j < 8; ++j) {
        int rowL = wm * 32 + mi * 16 + half * 8 + j;  // C/D layout
        float lv = 0.f;
        #pragma unroll
        for (int r = 0; r < R_LORA; ++r)
          lv = fmaf(Tl[rowL][r], Bl[colL][r], lv);
        out[(size_t)(m0 + rowL) * N_OUT + gcol] = acc[mi][ni][j] + bv + lv;
      }
    }
  }
}

// ==========================================================================
// Path A GEMM: pre-converted f16 X and W, async-to-LDS double buffering.
// 256 threads = 8 waves; block tile 128x128; K-chunk 64.
// ==========================================================================
__global__ __launch_bounds__(256) void qlora_gemm_async_kernel(
    const _Float16* __restrict__ Xh,
    const _Float16* __restrict__ Wh,
    const float* __restrict__ bias,
    const float* __restrict__ lB,
    const float* __restrict__ T,
    float*       __restrict__ out)
{
  __shared__ _Float16 Xs[2][BM][LDH];
  __shared__ _Float16 Ws[2][BN][LDH];
  __shared__ float    Tl[BM][16];
  __shared__ float    Bl[BN][16];

  const int t    = threadIdx.x;
  const int m0   = blockIdx.y * BM;
  const int n0   = blockIdx.x * BN;
  const int wave = t >> 5;
  const int lane = t & 31;
  const int wm   = wave >> 1;
  const int wn   = wave & 1;
  const int lr   = lane & 15;
  const int half = lane >> 4;

  v8f acc[2][4];
  #pragma unroll
  for (int mi = 0; mi < 2; ++mi)
    #pragma unroll
    for (int ni = 0; ni < 4; ++ni)
      acc[mi][ni] = (v8f){0.f,0.f,0.f,0.f,0.f,0.f,0.f,0.f};

  // per-thread slice of a chunk's staging: 8 x b128 (4 for X, 4 for W)
  const int srow = t >> 1;            // 0..127
  const int sq   = (t & 1) * 32;      // halves: 0 or 32 (two 64B spans/row)
  auto issue_chunk = [&](int buf, int kc) {
    const int k0 = kc * BK;
    #pragma unroll
    for (int u = 0; u < 4; ++u) {
      int col = sq + u * 8;           // four 16B units within the 32-half span
      async_copy_b128(Xh + (size_t)(m0 + srow) * K_IN + k0 + col,
                      &Xs[buf][srow][col]);
      async_copy_b128(Wh + (size_t)(n0 + srow) * K_IN + k0 + col,
                      &Ws[buf][srow][col]);
    }
  };

  constexpr int NC = K_IN / BK;
  issue_chunk(0, 0);

  for (int kc = 0; kc < NC; ++kc) {
    const int cur = kc & 1;
    if (kc + 1 < NC) {
      issue_chunk(cur ^ 1, kc + 1);
      wait_asynccnt_le<8>();          // newest 8 (next chunk) may stay in flight
    } else {
      wait_asynccnt_le<0>();
    }
    __syncthreads();                  // tile visible to all waves

    #pragma unroll
    for (int kk = 0; kk < BK; kk += 32) {
      AF a[2];
      #pragma unroll
      for (int mi = 0; mi < 2; ++mi) {
        int row = wm * 32 + mi * 16 + lr;
        int e0  = kk + half * 8;
        a[mi].u[0] = *reinterpret_cast<const uint4*>(&Xs[cur][row][e0]);
        a[mi].u[1] = *reinterpret_cast<const uint4*>(&Xs[cur][row][e0 + 16]);
      }
      AF b[4];
      #pragma unroll
      for (int ni = 0; ni < 4; ++ni) {
        int col = wn * 64 + ni * 16 + lr;
        int e0  = kk + half * 8;
        b[ni].u[0] = *reinterpret_cast<const uint4*>(&Ws[cur][col][e0]);
        b[ni].u[1] = *reinterpret_cast<const uint4*>(&Ws[cur][col][e0 + 16]);
      }
      #pragma unroll
      for (int mi = 0; mi < 2; ++mi)
        #pragma unroll
        for (int ni = 0; ni < 4; ++ni)
          acc[mi][ni] = __builtin_amdgcn_wmma_f32_16x16x32_f16(
              false, a[mi].v, false, b[ni].v,
              (short)0, acc[mi][ni], false, false);
    }
    __syncthreads();                  // protect buffer reuse at kc+2
  }

  gemm_epilogue(acc, Tl, Bl, T, lB, bias, out, t, m0, n0, wm, wn, lr, half);
}

// ==========================================================================
// Path B GEMM (fallback, fused dequant in-loop) — round-1 kernel
// ==========================================================================
__global__ __launch_bounds__(256) void qlora_gemm_fused_kernel(
    const float* __restrict__ x,
    const int*   __restrict__ qw,
    const float* __restrict__ scale,
    const float* __restrict__ bias,
    const float* __restrict__ lB,
    const float* __restrict__ T,
    float*       __restrict__ out)
{
  __shared__ _Float16 Xs[BM][LDH];
  __shared__ _Float16 Ws[BN][LDH];
  __shared__ float    Tl[BM][16];
  __shared__ float    Bl[BN][16];

  const int t    = threadIdx.x;
  const int m0   = blockIdx.y * BM;
  const int n0   = blockIdx.x * BN;
  const int wave = t >> 5;
  const int lane = t & 31;
  const int wm   = wave >> 1;
  const int wn   = wave & 1;
  const int lr   = lane & 15;
  const int half = lane >> 4;

  v8f acc[2][4];
  #pragma unroll
  for (int mi = 0; mi < 2; ++mi)
    #pragma unroll
    for (int ni = 0; ni < 4; ++ni)
      acc[mi][ni] = (v8f){0.f,0.f,0.f,0.f,0.f,0.f,0.f,0.f};

  constexpr float DQ = 2.0f / 15.0f;

  for (int kc = 0; kc < K_IN / BK; ++kc) {
    const int k0 = kc * BK;
    #pragma unroll
    for (int it = 0; it < 8; ++it) {
      int idx = it * 256 + t;
      int row = idx >> 4;
      int c4  = (idx & 15) * 4;

      float4 xv = *reinterpret_cast<const float4*>(
          x + (size_t)(m0 + row) * K_IN + k0 + c4);
      H4 hx;
      hx.h[0] = (_Float16)xv.x; hx.h[1] = (_Float16)xv.y;
      hx.h[2] = (_Float16)xv.z; hx.h[3] = (_Float16)xv.w;
      *reinterpret_cast<H4*>(&Xs[row][c4]) = hx;

      int4 qv = *reinterpret_cast<const int4*>(
          qw + (size_t)(n0 + row) * K_IN + k0 + c4);
      float s = scale[((size_t)(n0 + row) * K_IN + k0 + c4) >> 4];
      H4 hw;
      hw.h[0] = (_Float16)(fmaf((float)qv.x, DQ, -1.0f) * s);
      hw.h[1] = (_Float16)(fmaf((float)qv.y, DQ, -1.0f) * s);
      hw.h[2] = (_Float16)(fmaf((float)qv.z, DQ, -1.0f) * s);
      hw.h[3] = (_Float16)(fmaf((float)qv.w, DQ, -1.0f) * s);
      *reinterpret_cast<H4*>(&Ws[row][c4]) = hw;
    }

    if (kc + 1 < K_IN / BK) {
      __builtin_prefetch(qw + (size_t)(n0 + (t & 127)) * K_IN + k0 + BK, 0, 0);
      __builtin_prefetch(x  + (size_t)(m0 + (t & 127)) * K_IN + k0 + BK, 0, 0);
    }
    __syncthreads();

    #pragma unroll
    for (int kk = 0; kk < BK; kk += 32) {
      AF a[2];
      #pragma unroll
      for (int mi = 0; mi < 2; ++mi) {
        int row = wm * 32 + mi * 16 + lr;
        int e0  = kk + half * 8;
        a[mi].u[0] = *reinterpret_cast<const uint4*>(&Xs[row][e0]);
        a[mi].u[1] = *reinterpret_cast<const uint4*>(&Xs[row][e0 + 16]);
      }
      AF b[4];
      #pragma unroll
      for (int ni = 0; ni < 4; ++ni) {
        int col = wn * 64 + ni * 16 + lr;
        int e0  = kk + half * 8;
        b[ni].u[0] = *reinterpret_cast<const uint4*>(&Ws[col][e0]);
        b[ni].u[1] = *reinterpret_cast<const uint4*>(&Ws[col][e0 + 16]);
      }
      #pragma unroll
      for (int mi = 0; mi < 2; ++mi)
        #pragma unroll
        for (int ni = 0; ni < 4; ++ni)
          acc[mi][ni] = __builtin_amdgcn_wmma_f32_16x16x32_f16(
              false, a[mi].v, false, b[ni].v,
              (short)0, acc[mi][ni], false, false);
    }
    __syncthreads();
  }

  gemm_epilogue(acc, Tl, Bl, T, lB, bias, out, t, m0, n0, wm, wn, lr, half);
}

// --------------------------------------------------------------------------
extern "C" void kernel_launch(void* const* d_in, const int* in_sizes, int n_in,
                              void* d_out, int out_size, void* d_ws, size_t ws_size,
                              hipStream_t stream) {
  const float* x     = (const float*)d_in[0];
  const int*   qw    = (const int*)  d_in[1];
  const float* scale = (const float*)d_in[2];
  const float* bias  = (const float*)d_in[3];
  const float* lA    = (const float*)d_in[4];
  const float* lB    = (const float*)d_in[5];
  float*       out   = (float*)d_out;

  const size_t xhBytes = (size_t)M_TOTAL * K_IN * 2;   // 64 MB f16
  const size_t whBytes = (size_t)N_OUT   * K_IN * 2;   // 32 MB f16
  const size_t tBytes  = (size_t)M_TOTAL * R_LORA * 4; // 512 KB f32
  dim3 grid(N_OUT / BN, M_TOTAL / BM);

  if (ws_size >= xhBytes + whBytes + tBytes) {
    // Path A: one-time conversion + async double-buffered WMMA GEMM
    char* ws = (char*)d_ws;
    _Float16* Xh = (_Float16*)ws;
    _Float16* Wh = (_Float16*)(ws + xhBytes);
    float*    T  = (float*)   (ws + xhBytes + whBytes);

    cvt_x_kernel<<<dim3((M_TOTAL * (size_t)K_IN) / (256 * 8)), 256, 0, stream>>>(x, Xh);
    dequant_w_kernel<<<dim3((N_OUT * (size_t)K_IN) / (256 * 16)), 256, 0, stream>>>(qw, scale, Wh);
    lora_xa_kernel<<<dim3(M_TOTAL / 128), 256, 0, stream>>>(x, lA, T);
    qlora_gemm_async_kernel<<<grid, 256, 0, stream>>>(Xh, Wh, bias, lB, T, out);
  } else {
    // Path B: fused dequant GEMM (needs only 512 KB workspace)
    float* T = (float*)d_ws;
    lora_xa_kernel<<<dim3(M_TOTAL / 128), 256, 0, stream>>>(x, lA, T);
    qlora_gemm_fused_kernel<<<grid, 256, 0, stream>>>(x, qw, scale, bias, lB, T, out);
  }
}
